// Attention_25434796327784
// MI455X (gfx1250) — matrix operand order repaired and checked by reading, not compiled, not run on hardware
//
#include <hip/hip_runtime.h>
#include <hip/hip_bf16.h>

// ---------------------------------------------------------------------------
// Multi-head attention forward for MI455X (gfx1250), bf16 WMMA everywhere.
// B=4, N=2048, C=1024, H=16, D=64.
// ---------------------------------------------------------------------------

#define B_SZ   4
#define N_SZ   2048
#define C_SZ   1024
#define H_SZ   16
#define D_SZ   64
#define QKV_C  3072
#define SCALE_Q 0.125f          // 64^-0.5

typedef __attribute__((ext_vector_type(16))) __bf16 v16bf;
typedef __attribute__((ext_vector_type(8)))  __bf16 v8bf;
typedef __attribute__((ext_vector_type(8)))  float  v8f;

__device__ __forceinline__ v8f wmma_bf16(v16bf a, v16bf b, v8f c) {
  // (neg_a, A, neg_b, B, c_mod, C, reuse_a, reuse_b)
  return __builtin_amdgcn_wmma_f32_16x16x32_bf16(false, a, false, b, (short)0, c,
                                                 false, false);
}

__device__ __forceinline__ v8f zero8() {
  v8f z;
#pragma unroll
  for (int i = 0; i < 8; ++i) z[i] = 0.0f;
  return z;
}

// A-fragment (16x32 bf16): lane holds two contiguous 8-element runs.
__device__ __forceinline__ v16bf load_ab(const __bf16* p0, const __bf16* p1) {
  v8bf a = *(const v8bf*)p0;
  v8bf b = *(const v8bf*)p1;
  v16bf r;
#pragma unroll
  for (int i = 0; i < 8; ++i) { r[i] = a[i]; r[i + 8] = b[i]; }
  return r;
}

// ---------------------------------------------------------------------------
// 0) f32 -> bf16 conversion
// ---------------------------------------------------------------------------
__global__ void cvt_bf16_kernel(const float* __restrict__ in,
                                __bf16* __restrict__ out, int n) {
  int i = blockIdx.x * blockDim.x + threadIdx.x;
  if (i < n) out[i] = (__bf16)in[i];
}

// ---------------------------------------------------------------------------
// 1) QKV GEMM: qkv[m, o] = sum_c x[m,c] * w_qkv[o,c]   (q part scaled)
//    M=8192, N=3072, K=1024.  64x64 tile per wave, 4 waves per block.
// ---------------------------------------------------------------------------
__global__ __launch_bounds__(128)
void gemm_qkv_kernel(const __bf16* __restrict__ A, const __bf16* __restrict__ W,
                     __bf16* __restrict__ Cout) {
  const int lane = threadIdx.x & 31;
  const int wave = threadIdx.x >> 5;
  const int lo = lane & 15, hi = lane >> 4;
  const int m0 = blockIdx.x * 64;
  const int n0 = (blockIdx.y * 4 + wave) * 64;
  const int K = C_SZ;

  v8f acc[4][4];
#pragma unroll
  for (int i = 0; i < 4; ++i)
#pragma unroll
    for (int j = 0; j < 4; ++j) acc[i][j] = zero8();

  for (int k0 = 0; k0 < K; k0 += 32) {
    v16bf afr[4];
#pragma unroll
    for (int i = 0; i < 4; ++i) {
      const __bf16* p = A + (size_t)(m0 + i * 16 + lo) * K + k0 + hi * 16;
      afr[i] = load_ab(p, p + 8);
    }
#pragma unroll
    for (int j = 0; j < 4; ++j) {
      const __bf16* p = W + (size_t)(n0 + j * 16 + lo) * K + k0 + hi * 16;
      v16bf bfr = *(const v16bf*)p;
#pragma unroll
      for (int i = 0; i < 4; ++i) acc[i][j] = wmma_bf16(afr[i], bfr, acc[i][j]);
    }
  }

#pragma unroll
  for (int i = 0; i < 4; ++i)
#pragma unroll
    for (int j = 0; j < 4; ++j) {
      const int o = n0 + j * 16 + lo;
      const float s = (o < C_SZ) ? SCALE_Q : 1.0f;   // scale the q columns
#pragma unroll
      for (int r = 0; r < 8; ++r) {
        const int m = m0 + i * 16 + r + 8 * hi;
        Cout[(size_t)m * QKV_C + o] = (__bf16)(acc[i][j][r] * s);
      }
    }
}

// ---------------------------------------------------------------------------
// 2) Transpose V:  vT[b,h,d,n] = qkv[b,n, 2C + h*64 + d]
// ---------------------------------------------------------------------------
__global__ void vT_kernel(const __bf16* __restrict__ qkv,
                          __bf16* __restrict__ vT) {
  size_t i = (size_t)blockIdx.x * blockDim.x + threadIdx.x;
  if (i >= (size_t)B_SZ * H_SZ * D_SZ * N_SZ) return;
  const int n = (int)(i & (N_SZ - 1));
  const int d = (int)((i >> 11) & 63);
  const int h = (int)((i >> 17) & 15);
  const int b = (int)(i >> 21);
  vT[i] = qkv[((size_t)(b * N_SZ + n)) * QKV_C + 2 * C_SZ + h * D_SZ + d];
}

// ---------------------------------------------------------------------------
// 3) Flash-style attention. One wave owns a 16-query tile of one (b,h).
//    Keys processed in chunks of 256 with online softmax; PV computed as
//    O^T = V^T @ P^T so softmax rescale / 1/l are per-lane scalars.
//    Per-wave LDS: S (16x256 f32) + P (16x256 bf16) + stats = 24832 B.
// ---------------------------------------------------------------------------
#define ATT_WAVES 4
#define CHUNK 256
#define WAVE_LDS (16 * CHUNK * 4 + 16 * CHUNK * 2 + 256)

__global__ __launch_bounds__(128)
void attn_kernel(const __bf16* __restrict__ qkv, const __bf16* __restrict__ vT,
                 __bf16* __restrict__ out) {
  extern __shared__ char smem[];
  const int lane = threadIdx.x & 31;
  const int wave = threadIdx.x >> 5;
  const int lo = lane & 15, hi = lane >> 4;

  const int t  = blockIdx.x * ATT_WAVES + wave;  // global 16-query tile id
  const int qt = t & 127;                        // N/16 tiles
  const int h  = (t >> 7) & 15;
  const int b  = t >> 11;

  char* base   = smem + wave * WAVE_LDS;
  float*  S    = (float*)base;
  __bf16* P    = (__bf16*)(base + 16 * CHUNK * 4);
  float* stats = (float*)(base + 16 * CHUNK * 4 + 16 * CHUNK * 2);
  float* mrow = stats;            // [16] running row max
  float* lrow = stats + 16;       // [16] running row sum
  float* frow = stats + 32;       // [16] rescale factor this chunk

  if (lane < 16) { mrow[lane] = -3.0e38f; lrow[lane] = 0.0f; }

  // Q fragments (16 queries x 64 d) : two K=32 A-fragments, loaded once.
  const __bf16* qbase =
      qkv + (size_t)(b * N_SZ + qt * 16 + lo) * QKV_C + h * D_SZ;
  const v16bf aq0 = load_ab(qbase + hi * 16, qbase + hi * 16 + 8);
  const v16bf aq1 = load_ab(qbase + 32 + hi * 16, qbase + 32 + hi * 16 + 8);

  v8f oacc[4];                    // O^T : 4 d-tiles of 16x16 (d x query)
#pragma unroll
  for (int dt = 0; dt < 4; ++dt) oacc[dt] = zero8();

  const __bf16* vbase = vT + (size_t)((b * H_SZ + h) * D_SZ) * N_SZ;

  for (int kc = 0; kc < N_SZ; kc += CHUNK) {
    // ---- Phase A: S[16 x 256] = Q @ K^T for this key chunk -----------------
    for (int kt = 0; kt < CHUNK; kt += 16) {
      const __bf16* kp =
          qkv + (size_t)(b * N_SZ + kc + kt + lo) * QKV_C + C_SZ + h * D_SZ;
      v16bf b0 = *(const v16bf*)(kp + hi * 16);
      v16bf b1 = *(const v16bf*)(kp + 32 + hi * 16);
      v8f s = zero8();
      s = wmma_bf16(aq0, b0, s);
      s = wmma_bf16(aq1, b1, s);
#pragma unroll
      for (int r = 0; r < 8; ++r)
        S[(r + 8 * hi) * CHUNK + kt + lo] = s[r];
    }

    // ---- Phase B: online softmax update; emit P (bf16) ---------------------
    for (int r = 0; r < 16; ++r) {
      const float* Sr = S + r * CHUNK;
      float mx = -3.0e38f;
#pragma unroll
      for (int j = 0; j < CHUNK / 32; ++j) mx = fmaxf(mx, Sr[lane + j * 32]);
#pragma unroll
      for (int off = 16; off >= 1; off >>= 1)
        mx = fmaxf(mx, __shfl_xor(mx, off, 32));
      const float oldm = mrow[r];
      const float newm = fmaxf(oldm, mx);
      const float fs = __expf(oldm - newm);
      float sum = 0.0f;
#pragma unroll
      for (int j = 0; j < CHUNK / 32; ++j) {
        const int idx = lane + j * 32;
        const float e = __expf(Sr[idx] - newm);
        P[r * CHUNK + idx] = (__bf16)e;
        sum += e;
      }
#pragma unroll
      for (int off = 16; off >= 1; off >>= 1) sum += __shfl_xor(sum, off, 32);
      if (lane == 0) {
        mrow[r] = newm;
        lrow[r] = lrow[r] * fs + sum;
        frow[r] = fs;
      }
    }

    // rescale O accumulator (per-lane: column = query lo)
    const float fq = frow[lo];
#pragma unroll
    for (int dt = 0; dt < 4; ++dt)
#pragma unroll
      for (int r = 0; r < 8; ++r) oacc[dt][r] *= fq;

    // ---- Phase C: O^T += V^T @ P^T over this chunk -------------------------
    for (int ks = 0; ks < CHUNK; ks += 32) {
      // B = P^T (32 keys x 16 queries): contiguous reads from LDS P rows
      v16bf bp = *(const v16bf*)(P + lo * CHUNK + ks + hi * 16);
#pragma unroll
      for (int dt = 0; dt < 4; ++dt) {
        const __bf16* vp =
            vbase + (size_t)(dt * 16 + lo) * N_SZ + kc + ks + hi * 16;
        v16bf av = load_ab(vp, vp + 8);
        oacc[dt] = wmma_bf16(av, bp, oacc[dt]);
      }
    }
  }

  // ---- Epilogue: scale by 1/l, store [b, n, h*64+d] as bf16 ----------------
  const float linv = 1.0f / lrow[lo];
  const size_t obase = (size_t)(b * N_SZ + qt * 16 + lo) * C_SZ + h * D_SZ;
#pragma unroll
  for (int dt = 0; dt < 4; ++dt)
#pragma unroll
    for (int r = 0; r < 8; ++r) {
      const int d = dt * 16 + r + 8 * hi;
      out[obase + d] = (__bf16)(oacc[dt][r] * linv);
    }
}

// ---------------------------------------------------------------------------
// 4) Output projection: out[m, o] = ao[m,:] @ w_proj[o,:] + b_proj[o]  (f32)
// ---------------------------------------------------------------------------
__global__ __launch_bounds__(128)
void gemm_proj_kernel(const __bf16* __restrict__ A, const __bf16* __restrict__ W,
                      const float* __restrict__ bias, float* __restrict__ Out) {
  const int lane = threadIdx.x & 31;
  const int wave = threadIdx.x >> 5;
  const int lo = lane & 15, hi = lane >> 4;
  const int m0 = blockIdx.x * 64;
  const int n0 = (blockIdx.y * 4 + wave) * 64;
  const int K = C_SZ;

  v8f acc[4][4];
#pragma unroll
  for (int i = 0; i < 4; ++i)
#pragma unroll
    for (int j = 0; j < 4; ++j) acc[i][j] = zero8();

  for (int k0 = 0; k0 < K; k0 += 32) {
    v16bf afr[4];
#pragma unroll
    for (int i = 0; i < 4; ++i) {
      const __bf16* p = A + (size_t)(m0 + i * 16 + lo) * K + k0 + hi * 16;
      afr[i] = load_ab(p, p + 8);
    }
#pragma unroll
    for (int j = 0; j < 4; ++j) {
      const __bf16* p = W + (size_t)(n0 + j * 16 + lo) * K + k0 + hi * 16;
      v16bf bfr = *(const v16bf*)p;
#pragma unroll
      for (int i = 0; i < 4; ++i) acc[i][j] = wmma_bf16(afr[i], bfr, acc[i][j]);
    }
  }

#pragma unroll
  for (int i = 0; i < 4; ++i)
#pragma unroll
    for (int j = 0; j < 4; ++j) {
      const int o = n0 + j * 16 + lo;
      const float bb = bias[o];
#pragma unroll
      for (int r = 0; r < 8; ++r) {
        const int m = m0 + i * 16 + r + 8 * hi;
        Out[(size_t)m * C_SZ + o] = acc[i][j][r] + bb;
      }
    }
}

// ---------------------------------------------------------------------------
// Host-side launch
// ---------------------------------------------------------------------------
extern "C" void kernel_launch(void* const* d_in, const int* in_sizes, int n_in,
                              void* d_out, int out_size, void* d_ws, size_t ws_size,
                              hipStream_t stream) {
  const float* x      = (const float*)d_in[0];
  const float* w_qkv  = (const float*)d_in[1];
  const float* w_proj = (const float*)d_in[2];
  const float* b_proj = (const float*)d_in[3];
  float* out = (float*)d_out;

  char* ws = (char*)d_ws;
  // workspace layout (bytes)
  __bf16* xb     = (__bf16*)(ws);                  // 16,777,216
  __bf16* wqkvb  = (__bf16*)(ws + 16777216);       //  6,291,456
  __bf16* wprojb = (__bf16*)(ws + 23068672);       //  2,097,152
  __bf16* qkvb   = (__bf16*)(ws + 25165824);       // 50,331,648
  __bf16* vTb    = (__bf16*)(ws + 75497472);       // 16,777,216
  __bf16* aob    = (__bf16*)(ws + 92274688);       // 16,777,216  (total ~104MB)

  const int nx = B_SZ * N_SZ * C_SZ;       // 8,388,608
  const int nwq = 3 * C_SZ * C_SZ;         // 3,145,728
  const int nwp = C_SZ * C_SZ;             // 1,048,576

  cvt_bf16_kernel<<<nx / 256, 256, 0, stream>>>(x, xb, nx);
  cvt_bf16_kernel<<<nwq / 256, 256, 0, stream>>>(w_qkv, wqkvb, nwq);
  cvt_bf16_kernel<<<nwp / 256, 256, 0, stream>>>(w_proj, wprojb, nwp);

  // QKV GEMM: M=8192 (128 row-blocks of 64), N=3072 (12 * 4 waves * 64)
  gemm_qkv_kernel<<<dim3(128, 12), 128, 0, stream>>>(xb, wqkvb, qkvb);

  // V transpose
  vT_kernel<<<nx / 256, 256, 0, stream>>>(qkvb, vTb);

  // Attention: 8192 query tiles / 4 waves per block
  attn_kernel<<<2048, 128, ATT_WAVES * WAVE_LDS, stream>>>(qkvb, vTb, aob);

  // Output projection: N=1024 (4 * 4 waves * 64)
  gemm_proj_kernel<<<dim3(128, 4), 128, 0, stream>>>(aob, wprojb, b_proj, out);
}
